// UrbanWindViT_29214367548147
// MI455X (gfx1250) — matrix operand order repaired and checked by reading, not compiled, not run on hardware
//
#include <hip/hip_runtime.h>
#include <hip/hip_bf16.h>
#include <math.h>

// ---------------- WMMA types ----------------
typedef __attribute__((ext_vector_type(16))) _Float16 v16h;
typedef __attribute__((ext_vector_type(8)))  _Float16 v8h;
typedef __attribute__((ext_vector_type(8)))  float    v8f;

#define FLAG_RELU  1
#define FLAG_ACCUM 2

// CDNA5 async Global->LDS copy (ASYNCcnt-tracked DMA), per CDNA5 ISA.
// VDST = VGPR with wave-relative LDS byte offset (low 32 bits of a generic
// shared pointer), VADDR = 64-bit global address, GV mode.
__device__ __forceinline__ void async_copy_b128(unsigned ldsoff, const float* g) {
  asm volatile("global_load_async_to_lds_b128 %0, %1, off"
               :: "v"(ldsoff), "v"(g) : "memory");
}
__device__ __forceinline__ void wait_async_all() {
  asm volatile("s_wait_asynccnt 0x0" ::: "memory");
}
__device__ __forceinline__ void wait_async_8() {
  asm volatile("s_wait_asynccnt 0x8" ::: "memory");
}
__device__ __forceinline__ unsigned lds_off_u32(const void* p) {
  return (unsigned)(size_t)p;  // low 32 bits of shared flat addr = LDS offset
}

__device__ __forceinline__ v16h frag_from_lds(const _Float16* p0, const _Float16* p1) {
  v8h lo = *(const v8h*)p0;   // ds_load_b128
  v8h hi = *(const v8h*)p1;   // ds_load_b128
  v16h r;
#pragma unroll
  for (int e = 0; e < 8; ++e) { r[e] = lo[e]; r[8 + e] = hi[e]; }
  return r;
}

// ---------------------------------------------------------------------------
// f16-WMMA GEMM with LDS-staged tiles and async double-buffered DMA.
// C[M,N] = A[M,K] @ B[K,N] (+bias)(+=C)(relu). B row-major, or [N,K] if transB.
// Block = 128 threads (4 waves) -> 64x64 C tile; each wave computes 32x32
// (4 x v_wmma_f32_16x16x32_f16 per k-step with A/B fragment reuse).
// K % 32 == 0 at every call site; M/N edges handled by clamp+zero+guard.
// ---------------------------------------------------------------------------
__global__ void gemm_wmma_kernel(const float* __restrict__ A, int lda,
                                 const float* __restrict__ B, int ldb,
                                 float* __restrict__ C, int ldc,
                                 const float* __restrict__ bias,
                                 int M, int N, int K, int flags, int transB) {
  __shared__ __align__(16) _Float16 As16[64 * 32];      // [row][k]
  __shared__ __align__(16) _Float16 Bs16[64 * 32];      // [n][k] (col-major)
  __shared__ __align__(16) float    Araw[2][64 * 32];   // raw f32 double buffer
  __shared__ __align__(16) float    Braw[2][64 * 32];

  const int t = threadIdx.x;
  const int wave = t >> 5, lane = t & 31;
  const int wm = wave >> 1, wn = wave & 1;
  const int hl = lane >> 4, l15 = lane & 15;
  const int m0b = blockIdx.x * 64;
  const int n0b = blockIdx.y * 64;
  const int nsteps = K >> 5;

  // --- async issue of one 64x32 A tile + 32x64 (or 64x32) B tile ---
  auto issue_tile = [&](int buf, int kk) {
#pragma unroll
    for (int j = 0; j < 4; ++j) {               // A: 512 x b128
      int idx = t + j * 128;
      int row = idx >> 3, kq = (idx & 7) << 2;
      int rg = m0b + row; if (rg > M - 1) rg = M - 1;
      async_copy_b128(lds_off_u32(&Araw[buf][idx << 2]),
                      A + (size_t)rg * lda + kk + kq);
    }
    if (transB) {
#pragma unroll
      for (int j = 0; j < 4; ++j) {             // B[N,K]: raw [n][k]
        int idx = t + j * 128;
        int n = idx >> 3, kq = (idx & 7) << 2;
        int ng = n0b + n; if (ng > N - 1) ng = N - 1;
        async_copy_b128(lds_off_u32(&Braw[buf][idx << 2]),
                        B + (size_t)ng * ldb + kk + kq);
      }
    } else {
#pragma unroll
      for (int j = 0; j < 4; ++j) {             // B[K,N]: raw [k][n]
        int idx = t + j * 128;
        int k = idx >> 4, nq = (idx & 15) << 2;
        int ng = n0b + nq; if (ng > N - 4) ng = N - 4;   // N % 4 == 0 always
        async_copy_b128(lds_off_u32(&Braw[buf][idx << 2]),
                        B + (size_t)(kk + k) * ldb + ng);
      }
    }
  };

  issue_tile(0, 0);  // prologue

  v8f acc00 = {}, acc01 = {}, acc10 = {}, acc11 = {};

  for (int s = 0; s < nsteps; ++s) {
    const int cur = s & 1;
    const bool more = (s + 1) < nsteps;
    if (more) issue_tile(cur ^ 1, (s + 1) << 5);
    if (more) wait_async_8(); else wait_async_all();   // in-order retire
    __syncthreads();                                   // raw tile visible to all

    // --- convert raw f32 -> f16 tiles (once per block), zero-fill OOB ---
    const float* Ar = Araw[cur];
    const float* Br = Braw[cur];
#pragma unroll
    for (int j = 0; j < 16; ++j) {
      int idx = t + j * 128;                 // 2048 elements each
      {
        int row = idx >> 5;
        As16[idx] = (m0b + row < M) ? (_Float16)Ar[idx] : (_Float16)0.f;
      }
      {
        int n = idx >> 5, k = idx & 31;
        float v = transB ? Br[idx] : Br[k * 64 + n];
        Bs16[n * 32 + k] = (n0b + n < N) ? (_Float16)v : (_Float16)0.f;
      }
    }
    __syncthreads();

    // --- fragments from LDS (2 x ds_load_b128 each) ---
    const int r0 = wm * 32 + l15;
    const int c0 = wn * 32 + l15;
    v16h a0 = frag_from_lds(&As16[r0 * 32 + hl * 8],        &As16[r0 * 32 + 16 + hl * 8]);
    v16h a1 = frag_from_lds(&As16[(r0 + 16) * 32 + hl * 8], &As16[(r0 + 16) * 32 + 16 + hl * 8]);
    v16h b0 = frag_from_lds(&Bs16[c0 * 32 + hl * 8],        &Bs16[c0 * 32 + 16 + hl * 8]);
    v16h b1 = frag_from_lds(&Bs16[(c0 + 16) * 32 + hl * 8], &Bs16[(c0 + 16) * 32 + 16 + hl * 8]);

    acc00 = __builtin_amdgcn_wmma_f32_16x16x32_f16(false, a0, false, b0, (short)0, acc00, false, false);
    acc01 = __builtin_amdgcn_wmma_f32_16x16x32_f16(false, a0, false, b1, (short)0, acc01, false, false);
    acc10 = __builtin_amdgcn_wmma_f32_16x16x32_f16(false, a1, false, b0, (short)0, acc10, false, false);
    acc11 = __builtin_amdgcn_wmma_f32_16x16x32_f16(false, a1, false, b1, (short)0, acc11, false, false);
    __syncthreads();   // frags consumed before next convert overwrites tiles
  }

  // --- epilogue: C/D layout row = r + 8*hl, col = lane&15 ---
#pragma unroll
  for (int rm = 0; rm < 2; ++rm) {
#pragma unroll
    for (int cn = 0; cn < 2; ++cn) {
      const v8f& acc = rm == 0 ? (cn == 0 ? acc00 : acc01)
                               : (cn == 0 ? acc10 : acc11);
      int coln = n0b + wn * 32 + cn * 16 + l15;
      if (coln < N) {
#pragma unroll
        for (int r = 0; r < 8; ++r) {
          int row = m0b + wm * 32 + rm * 16 + r + hl * 8;
          if (row < M) {
            float v = acc[r];
            if (bias) v += bias[coln];
            float* cp = C + (size_t)row * ldc + coln;
            if (flags & FLAG_ACCUM) v += *cp;
            if (flags & FLAG_RELU) v = fmaxf(v, 0.f);
            *cp = v;
          }
        }
      }
    }
  }
}

// ---------------------------------------------------------------------------
// Stage 1: per-grid-cell top-k neighbor encode (2 scales).
// ---------------------------------------------------------------------------
__global__ void encode_kernel(const float* __restrict__ grid,
                              const float* __restrict__ pts,
                              const float* __restrict__ w1,
                              const float* __restrict__ b1,
                              const float* __restrict__ w2,
                              const float* __restrict__ b2,
                              float* __restrict__ gfeat, int npts) {
  int cell = blockIdx.x;
  int sc = blockIdx.y;
  int t = threadIdx.x;
  const int K = sc ? 32 : 16;
  const float r2 = sc ? (0.3f * 0.3f) : (0.15f * 0.15f);
  const float* W1 = w1 + sc * 64;
  const float* B1 = b1 + sc * 32;
  const float* W2 = w2 + sc * 2048;
  const float* B2 = b2 + sc * 64;
  float gx = grid[cell * 2], gy = grid[cell * 2 + 1];

  __shared__ float pd[256];
  __shared__ int   pidx[256];
  __shared__ float seld[32];
  __shared__ int   seli[32];
  __shared__ float lastD_s;
  __shared__ int   lastI_s;
  __shared__ float hb[32];
  __shared__ float pooled[64];
  __shared__ int   anyv;

  if (t == 0) { lastD_s = -1e30f; lastI_s = -1; anyv = 0; }
  if (t < 64) pooled[t] = -1e30f;
  __syncthreads();

  for (int it = 0; it < K; ++it) {
    float lastD = lastD_s; int lastI = lastI_s;
    float bd = 1e30f; int bi = 0x7fffffff;
    for (int p = t; p < npts; p += 256) {
      float dx = gx - pts[p * 2], dy = gy - pts[p * 2 + 1];
      float d2 = dx * dx + dy * dy;
      bool after = (d2 > lastD) || (d2 == lastD && p > lastI);
      if (after && (d2 < bd || (d2 == bd && p < bi))) { bd = d2; bi = p; }
    }
    pd[t] = bd; pidx[t] = bi;
    __syncthreads();
    for (int s = 128; s > 0; s >>= 1) {
      if (t < s) {
        float od = pd[t + s]; int oi = pidx[t + s];
        if (od < pd[t] || (od == pd[t] && oi < pidx[t])) { pd[t] = od; pidx[t] = oi; }
      }
      __syncthreads();
    }
    if (t == 0) { seld[it] = pd[0]; seli[it] = pidx[0]; lastD_s = pd[0]; lastI_s = pidx[0]; }
    __syncthreads();
  }

  for (int j = 0; j < K; ++j) {
    float d2 = seld[j]; int p = seli[j];
    if (d2 < r2 && p < npts) {           // uniform (shared values)
      if (t == 0) anyv = 1;
      float rx = pts[p * 2] - gx, ry = pts[p * 2 + 1] - gy;
      if (t < 32) hb[t] = fmaxf(0.f, rx * W1[t] + ry * W1[32 + t] + B1[t]);
      __syncthreads();
      if (t < 64) {
        float a = B2[t];
        for (int i = 0; i < 32; ++i) a += hb[i] * W2[i * 64 + t];
        pooled[t] = fmaxf(pooled[t], a);
      }
      __syncthreads();
    }
  }
  __syncthreads();
  if (t < 64) gfeat[(size_t)cell * 128 + sc * 64 + t] = anyv ? pooled[t] : 0.f;
}

// ---------------- ViT helper kernels ----------------
__global__ void rmsnorm_kernel(const float* __restrict__ X,
                               const float* __restrict__ w,
                               float* __restrict__ Y) {
  int row = blockIdx.x, t = threadIdx.x;  // D = 256 = blockDim
  __shared__ float red[256];
  float x = X[(size_t)row * 256 + t];
  red[t] = x * x;
  __syncthreads();
  for (int s = 128; s > 0; s >>= 1) { if (t < s) red[t] += red[t + s]; __syncthreads(); }
  float n = rsqrtf(red[0] * (1.0f / 256.0f) + 1e-6f);
  Y[(size_t)row * 256 + t] = x * n * w[t];
}

__global__ void patchify_kernel(const float* __restrict__ gfeatp, float* __restrict__ pin) {
  int tid = blockIdx.x * 256 + threadIdx.x;
  if (tid >= 1024 * 1024) return;
  int tok = tid >> 10, f = tid & 1023;
  int c = f >> 2, ph = (f >> 1) & 1, pw = f & 1;
  int gy = (tok >> 5) * 2 + ph, gx = (tok & 31) * 2 + pw;
  pin[tid] = gfeatp[(size_t)(gy * 64 + gx) * 256 + c];
}

__global__ void unpatch_kernel(const float* __restrict__ up, float* __restrict__ proc) {
  int tid = blockIdx.x * 256 + threadIdx.x;
  if (tid >= 1024 * 1024) return;
  int tok = tid >> 10, f = tid & 1023;
  int c = f >> 2, ph = (f >> 1) & 1, pw = f & 1;
  int gy = (tok >> 5) * 2 + ph, gx = (tok & 31) * 2 + pw;
  proc[(size_t)c * 4096 + gy * 64 + gx] = up[tid];
}

__global__ void rope_kernel(float* __restrict__ qkv) {
  int tid = blockIdx.x * blockDim.x + threadIdx.x;  // T*NH*8 = 65536
  if (tid >= 1024 * 8 * 8) return;
  int i = tid & 7, h = (tid >> 3) & 7, t = tid >> 6;
  float tw = (float)(t & 31), th = (float)(t >> 5);
  float inv = __powf(10000.f, -(float)i * 0.125f);
  float cxv = __cosf(tw * inv), sxv = __sinf(tw * inv);
  float cyv = __cosf(th * inv), syv = __sinf(th * inv);
  for (int s = 0; s < 2; ++s) {  // q then k
    float* base = qkv + (size_t)t * 768 + s * 256 + h * 32;
    float x0 = base[2 * i], x1 = base[2 * i + 1];
    base[2 * i]     = x0 * cxv - x1 * sxv;
    base[2 * i + 1] = x0 * sxv + x1 * cxv;
    float y0 = base[16 + 2 * i], y1 = base[16 + 2 * i + 1];
    base[16 + 2 * i]     = y0 * cyv - y1 * syv;
    base[16 + 2 * i + 1] = y0 * syv + y1 * cyv;
  }
}

__global__ void softmax_kernel(float* __restrict__ S, int T, float scale) {
  int row = blockIdx.x, t = threadIdx.x;
  float* r = S + (size_t)row * T;
  __shared__ float red[256];
  float m = -1e30f;
  for (int i = t; i < T; i += 256) m = fmaxf(m, r[i] * scale);
  red[t] = m; __syncthreads();
  for (int s = 128; s > 0; s >>= 1) { if (t < s) red[t] = fmaxf(red[t], red[t + s]); __syncthreads(); }
  m = red[0]; __syncthreads();
  float sum = 0.f;
  for (int i = t; i < T; i += 256) { float e = __expf(r[i] * scale - m); r[i] = e; sum += e; }
  red[t] = sum; __syncthreads();
  for (int s = 128; s > 0; s >>= 1) { if (t < s) red[t] += red[t + s]; __syncthreads(); }
  float invs = 1.0f / red[0];
  for (int i = t; i < T; i += 256) r[i] *= invs;
}

__global__ void silu_gate_kernel(float* __restrict__ g, const float* __restrict__ v, int n) {
  int i = blockIdx.x * 256 + threadIdx.x;
  if (i < n) { float x = g[i]; g[i] = (x / (1.f + __expf(-x))) * v[i]; }
}

// ---------------- Query path ----------------
__device__ __forceinline__ void query_xy(const float* qpos, int q, float& xn, float& yn) {
  float qx = qpos[(size_t)q * 2], qy = qpos[(size_t)q * 2 + 1];
  xn = fminf(fmaxf(2.f * (qx + 2.0f) / 6.0f - 1.f, -1.f), 1.f);
  yn = fminf(fmaxf(2.f * (qy + 1.5f) / 3.0f - 1.f, -1.f), 1.f);
}

__global__ void sample_kernel(const float* __restrict__ proc,
                              const float* __restrict__ qpos,
                              float* __restrict__ ctx, int q0, int nq) {
  int tid = blockIdx.x * 256 + threadIdx.x;
  if (tid >= nq * 256) return;
  int q = tid >> 8, c = tid & 255;
  float xn, yn; query_xy(qpos, q0 + q, xn, yn);
  float px = (xn + 1.f) * 0.5f * 63.f, py = (yn + 1.f) * 0.5f * 63.f;
  float x0f = fminf(fmaxf(floorf(px), 0.f), 63.f);
  float y0f = fminf(fmaxf(floorf(py), 0.f), 63.f);
  int x0 = (int)x0f, y0 = (int)y0f;
  int x1 = min(x0 + 1, 63), y1 = min(y0 + 1, 63);
  float wx = px - x0f, wy = py - y0f;
  const float* f = proc + (size_t)c * 4096;
  float v = f[y0 * 64 + x0] * (1.f - wx) * (1.f - wy)
          + f[y0 * 64 + x1] * wx * (1.f - wy)
          + f[y1 * 64 + x0] * (1.f - wx) * wy
          + f[y1 * 64 + x1] * wx * wy;
  ctx[(size_t)q * 768 + c] = v;
}

__global__ void pin_kernel(const float* __restrict__ qpos,
                           const float* __restrict__ uinf,
                           const float* __restrict__ sdf,
                           const float* __restrict__ sgrad,
                           float* __restrict__ pin, int q0, int nq) {
  int q = blockIdx.x * 256 + threadIdx.x;
  if (q >= nq) return;
  float xn, yn; query_xy(qpos, q0 + q, xn, yn);
  float* p = pin + (size_t)q * 64;
  p[0] = xn; p[1] = yn;
  const float PI = 3.14159265358979323846f;
  for (int j = 0; j < 10; ++j) {
    float fr = PI * (float)(1 << j);
    p[2 + j]  = __sinf(xn * fr);
    p[12 + j] = __cosf(xn * fr);
    p[22 + j] = __sinf(yn * fr);
    p[32 + j] = __cosf(yn * fr);
  }
  int g = q0 + q;
  p[42] = uinf[(size_t)g * 2]; p[43] = uinf[(size_t)g * 2 + 1];
  p[44] = sdf[g];
  p[45] = sgrad[(size_t)g * 2]; p[46] = sgrad[(size_t)g * 2 + 1];
  for (int j = 47; j < 64; ++j) p[j] = 0.f;
}

__global__ void w1pad_kernel(const float* __restrict__ w1, float* __restrict__ w1p) {
  int tid = blockIdx.x * 256 + threadIdx.x;  // 64*256
  if (tid >= 64 * 256) return;
  int i = tid >> 8, j = tid & 255;
  w1p[tid] = (i < 47) ? w1[i * 256 + j] : 0.f;
}

__global__ void pred2_kernel(const float* __restrict__ hp,
                             const float* __restrict__ w,
                             const float* __restrict__ b,
                             float* __restrict__ out, int q0, int nq) {
  int tid = blockIdx.x * 256 + threadIdx.x;
  if (tid >= nq * 4) return;
  int q = tid >> 2, n = tid & 3;
  const float* r = hp + (size_t)q * 256;
  float acc = b[n];
  for (int k = 0; k < 256; ++k) acc += r[k] * w[k * 4 + n];
  out[(size_t)(q0 + q) * 4 + n] = acc;
}

// ---------------------------------------------------------------------------
static inline void gemm(const float* A, int lda, const float* B, int ldb,
                        float* C, int ldc, const float* bias,
                        int M, int N, int K, int flags, int transB,
                        hipStream_t s) {
  dim3 g((M + 63) / 64, (N + 63) / 64), b(128);
  gemm_wmma_kernel<<<g, b, 0, s>>>(A, lda, B, ldb, C, ldc, bias, M, N, K, flags, transB);
}

extern "C" void kernel_launch(void* const* d_in, const int* in_sizes, int n_in,
                              void* d_out, int out_size, void* d_ws, size_t ws_size,
                              hipStream_t stream) {
  (void)in_sizes; (void)n_in; (void)out_size; (void)ws_size;
  const float* points     = (const float*)d_in[0];
  const float* grid       = (const float*)d_in[1];
  const float* query_pos  = (const float*)d_in[2];
  const float* query_uinf = (const float*)d_in[3];
  const float* query_sdf  = (const float*)d_in[4];
  const float* query_grad = (const float*)d_in[5];
  const float* enc_w1     = (const float*)d_in[6];
  const float* enc_b1     = (const float*)d_in[7];
  const float* enc_w2     = (const float*)d_in[8];
  const float* enc_b2     = (const float*)d_in[9];
  const float* inproj_w   = (const float*)d_in[10];
  const float* inproj_b   = (const float*)d_in[11];
  const float* patch_w    = (const float*)d_in[12];
  const float* patch_b    = (const float*)d_in[13];
  const float* norm1_w    = (const float*)d_in[14];
  const float* qkv_w      = (const float*)d_in[15];
  const float* attnproj_w = (const float*)d_in[16];
  const float* norm2_w    = (const float*)d_in[17];
  const float* gate_w     = (const float*)d_in[18];
  const float* value_w    = (const float*)d_in[19];
  const float* ffnout_w   = (const float*)d_in[20];
  const float* unpatch_w  = (const float*)d_in[21];
  const float* unpatch_b  = (const float*)d_in[22];
  const float* posmlp_w1  = (const float*)d_in[23];
  const float* posmlp_b1  = (const float*)d_in[24];
  const float* posmlp_w2  = (const float*)d_in[25];
  const float* posmlp_b2  = (const float*)d_in[26];
  const float* pred_w1    = (const float*)d_in[27];
  const float* pred_b1    = (const float*)d_in[28];
  const float* pred_w2    = (const float*)d_in[29];
  const float* pred_b2    = (const float*)d_in[30];
  float* out = (float*)d_out;

  // ---- workspace bump allocator (~145 MB with 20k-query chunking) ----
  char* ws = (char*)d_ws;
  size_t off = 0;
  auto alloc = [&](size_t bytes) -> float* {
    float* p = (float*)(ws + off);
    off += (bytes + 255) & ~(size_t)255;
    return p;
  };
  const int NPTS = 20000, NQ = 100000, QCH = 20000;
  float* gfeat   = alloc((size_t)4096 * 128 * 4);
  float* gfeatp  = alloc((size_t)4096 * 256 * 4);
  float* patchin = alloc((size_t)1024 * 1024 * 4);
  float* tok     = alloc((size_t)1024 * 256 * 4);
  float* hbuf    = alloc((size_t)1024 * 256 * 4);
  float* qkv     = alloc((size_t)1024 * 768 * 4);
  float* score   = alloc((size_t)1024 * 1024 * 4);
  float* obuf    = alloc((size_t)1024 * 256 * 4);
  float* gbuf    = alloc((size_t)1024 * 1024 * 4);
  float* vbuf    = alloc((size_t)1024 * 1024 * 4);
  float* up      = alloc((size_t)1024 * 1024 * 4);
  float* proc    = alloc((size_t)256 * 4096 * 4);
  float* w1pad   = alloc((size_t)64 * 256 * 4);
  float* pinb    = alloc((size_t)QCH * 64 * 4);
  float* ctx     = alloc((size_t)QCH * 768 * 4);
  float* hq      = alloc((size_t)QCH * 256 * 4);
  float* hp      = alloc((size_t)QCH * 256 * 4);

  // ---- Stage 1: multiscale neighbor encode -> gfeat (4096 x 128) ----
  encode_kernel<<<dim3(4096, 2), 256, 0, stream>>>(grid, points, enc_w1, enc_b1,
                                                   enc_w2, enc_b2, gfeat, NPTS);
  gemm(gfeat, 128, inproj_w, 256, gfeatp, 256, inproj_b, 4096, 256, 128, 0, 0, stream);

  // ---- Stage 2: patchify + ViT ----
  patchify_kernel<<<4096, 256, 0, stream>>>(gfeatp, patchin);
  gemm(patchin, 1024, patch_w, 256, tok, 256, patch_b, 1024, 256, 1024, 0, 0, stream);

  const float att_scale = 0.17677669529663687f;  // 1/sqrt(32)
  for (int l = 0; l < 5; ++l) {
    rmsnorm_kernel<<<1024, 256, 0, stream>>>(tok, norm1_w + l * 256, hbuf);
    gemm(hbuf, 256, qkv_w + (size_t)l * 256 * 768, 768, qkv, 768, nullptr,
         1024, 768, 256, 0, 0, stream);
    rope_kernel<<<256, 256, 0, stream>>>(qkv);
    for (int h = 0; h < 8; ++h) {
      // scores = q_h (1024x32) @ k_h^T  [B stored N x K -> transB]
      gemm(qkv + h * 32, 768, qkv + 256 + h * 32, 768, score, 1024, nullptr,
           1024, 1024, 32, 0, 1, stream);
      softmax_kernel<<<1024, 256, 0, stream>>>(score, 1024, att_scale);
      // o_h = att (1024x1024) @ v_h (1024x32)
      gemm(score, 1024, qkv + 512 + h * 32, 768, obuf + h * 32, 256, nullptr,
           1024, 32, 1024, 0, 0, stream);
    }
    gemm(obuf, 256, attnproj_w + (size_t)l * 256 * 256, 256, tok, 256, nullptr,
         1024, 256, 256, FLAG_ACCUM, 0, stream);
    rmsnorm_kernel<<<1024, 256, 0, stream>>>(tok, norm2_w + l * 256, hbuf);
    gemm(hbuf, 256, gate_w + (size_t)l * 256 * 1024, 1024, gbuf, 1024, nullptr,
         1024, 1024, 256, 0, 0, stream);
    gemm(hbuf, 256, value_w + (size_t)l * 256 * 1024, 1024, vbuf, 1024, nullptr,
         1024, 1024, 256, 0, 0, stream);
    silu_gate_kernel<<<4096, 256, 0, stream>>>(gbuf, vbuf, 1024 * 1024);
    gemm(gbuf, 1024, ffnout_w + (size_t)l * 1024 * 256, 256, tok, 256, nullptr,
         1024, 256, 1024, FLAG_ACCUM, 0, stream);
  }

  // ---- Stage 3: unpatch -> proc (256 x 64 x 64) ----
  gemm(tok, 256, unpatch_w, 1024, up, 1024, unpatch_b, 1024, 1024, 256, 0, 0, stream);
  unpatch_kernel<<<4096, 256, 0, stream>>>(up, proc);

  // ---- Stage 4: query decoder (chunked, 5 x 20000 queries) ----
  w1pad_kernel<<<64, 256, 0, stream>>>(posmlp_w1, w1pad);
  for (int q0 = 0; q0 < NQ; q0 += QCH) {
    sample_kernel<<<QCH, 256, 0, stream>>>(proc, query_pos, ctx, q0, QCH);
    pin_kernel<<<(QCH + 255) / 256, 256, 0, stream>>>(query_pos, query_uinf,
                                                      query_sdf, query_grad,
                                                      pinb, q0, QCH);
    gemm(pinb, 64, w1pad, 256, hq, 256, posmlp_b1, QCH, 256, 64, FLAG_RELU, 0, stream);
    gemm(hq, 256, posmlp_w2, 512, ctx + 256, 768, posmlp_b2, QCH, 512, 256, 0, 0, stream);
    gemm(ctx, 768, pred_w1, 256, hp, 256, pred_b1, QCH, 256, 768, FLAG_RELU, 0, stream);
    pred2_kernel<<<(QCH * 4 + 255) / 256, 256, 0, stream>>>(hp, pred_w2, pred_b2,
                                                            out, q0, QCH);
  }
}
